// SNNNet_53145925320929
// MI455X (gfx1250) — matrix operand order repaired and checked by reading, not compiled, not run on hardware
//
#include <hip/hip_runtime.h>

// ---------------------------------------------------------------------------
// SNN (2-layer LIF) for MI455X / gfx1250.
//   Phase 0: convert x, W1 to bf16 (W1 padded 1000->1024 rows with zeros).
//   Phase 1: cur1 = x @ W1^T + b1 via v_wmma_f32_16x16x32_bf16 (fp32 accum).
//   Phase 2: 30-step LIF recurrence, one wave32 per batch row, state fully
//            register-resident (cur1/m1: 32 VGPRs each per lane), W2 in LDS,
//            wave32 butterfly reduction for the 2-wide layer-2 current.
// ---------------------------------------------------------------------------

typedef __attribute__((ext_vector_type(16))) __bf16 v16bf;
typedef __attribute__((ext_vector_type(8)))  __bf16 v8bf;
typedef __attribute__((ext_vector_type(4)))  __bf16 v4bf;
typedef __attribute__((ext_vector_type(8)))  float  v8f;

#define BETA   0.8f
#define THR    1.0f
#define TSTEPS 30
#define BROWS  16384
#define DIN    1024
#define H      1000
#define HPAD   1024

// round-to-nearest-even fp32 -> bf16 (storage only)
__device__ __forceinline__ __bf16 f2bf(float f) {
  unsigned u = __builtin_bit_cast(unsigned, f);
  unsigned r = u + 0x7FFFu + ((u >> 16) & 1u);
  unsigned short h = (unsigned short)(r >> 16);
  return __builtin_bit_cast(__bf16, h);
}

// ---------------- Phase 0: bf16 conversion ----------------
__global__ void cvt_x_kernel(const float* __restrict__ x, __bf16* __restrict__ xb) {
  int i4 = blockIdx.x * blockDim.x + threadIdx.x;   // 4 elems / thread
  float4 v = ((const float4*)x)[i4];
  v4bf o;
  o[0] = f2bf(v.x); o[1] = f2bf(v.y); o[2] = f2bf(v.z); o[3] = f2bf(v.w);
  ((v4bf*)xb)[i4] = o;
}

__global__ void cvt_w1_kernel(const float* __restrict__ W1, __bf16* __restrict__ w1b) {
  int i4 = blockIdx.x * blockDim.x + threadIdx.x;   // dest padded [1024][1024]
  int d  = i4 * 4;
  v4bf o;
  if (d < H * DIN) {                                // rows share width -> same flat idx
    float4 v = *(const float4*)(W1 + d);
    o[0] = f2bf(v.x); o[1] = f2bf(v.y); o[2] = f2bf(v.z); o[3] = f2bf(v.w);
  } else {                                          // zero-pad rows 1000..1023
    __bf16 z = __builtin_bit_cast(__bf16, (unsigned short)0);
    o[0] = z; o[1] = z; o[2] = z; o[3] = z;
  }
  ((v4bf*)w1b)[i4] = o;
}

// ---------------- Phase 1: bf16 WMMA GEMM ----------------
// Block = 256 threads = 8 waves, wave grid 4(M) x 2(N).
// Wave tile: 32(M) x 64(N) = 2x4 WMMA 16x16 tiles, K-loop step 32.
// Block tile: 128 x 128. Grid: (1024/128, 16384/128) = (8, 128).
__global__ __launch_bounds__(256)
void gemm_bf16_wmma_kernel(const __bf16* __restrict__ xb,
                           const __bf16* __restrict__ w1b,
                           const float*  __restrict__ b1,
                           float*        __restrict__ cur1) {
  const int lane  = threadIdx.x & 31;
  const int wave  = threadIdx.x >> 5;
  const int waveM = wave & 3;
  const int waveN = wave >> 2;
  const int mBase = blockIdx.y * 128 + waveM * 32;
  const int nBase = blockIdx.x * 128 + waveN * 64;
  const int l16   = lane & 15;
  const int lHi   = lane >> 4;

  // Per-lane row pointers (contiguous along K in memory).
  const __bf16* aRow[2];
  const __bf16* bRow[4];
#pragma unroll
  for (int mi = 0; mi < 2; ++mi)
    aRow[mi] = xb + (size_t)(mBase + mi * 16 + l16) * DIN;
#pragma unroll
  for (int ni = 0; ni < 4; ++ni)
    bRow[ni] = w1b + (size_t)(nBase + ni * 16 + l16) * DIN;

  // ISA fragment layouts (16-bit, wave32):
  //  A 16x32: lanes 0-15 K in {0..7, 16..23}, lanes 16-31 K in {8..15, 24..31}
  //  B 32x16: lanes 0-15 K in {0..7, 16..23}, lanes 16-31 K in {8..15, 24..31}
  const int aK = lHi * 8;
  const int bK = lHi * 8;

  v8f zero = {};
  v8f c[2][4];
#pragma unroll
  for (int mi = 0; mi < 2; ++mi)
#pragma unroll
    for (int ni = 0; ni < 4; ++ni) c[mi][ni] = zero;

  for (int kk = 0; kk < DIN; kk += 32) {
    if (kk + 32 < DIN) {                 // CDNA5 global_prefetch_b8 for next K tile
      __builtin_prefetch(aRow[0] + kk + 32 + aK, 0, 3);
      __builtin_prefetch(bRow[0] + kk + 32 + bK, 0, 3);
    }
    v16bf a[2], b[4];
#pragma unroll
    for (int mi = 0; mi < 2; ++mi) {
      v8bf lo = *(const v8bf*)(aRow[mi] + kk + aK);
      v8bf hi = *(const v8bf*)(aRow[mi] + kk + aK + 16);
      a[mi] = __builtin_shufflevector(lo, hi, 0,1,2,3,4,5,6,7,8,9,10,11,12,13,14,15);
    }
#pragma unroll
    for (int ni = 0; ni < 4; ++ni) {
      v8bf lo = *(const v8bf*)(bRow[ni] + kk + bK);
      v8bf hi = *(const v8bf*)(bRow[ni] + kk + bK + 16);
      b[ni] = __builtin_shufflevector(lo, hi, 0,1,2,3,4,5,6,7,8,9,10,11,12,13,14,15);
    }
#pragma unroll
    for (int mi = 0; mi < 2; ++mi)
#pragma unroll
      for (int ni = 0; ni < 4; ++ni)
        c[mi][ni] = __builtin_amdgcn_wmma_f32_16x16x32_bf16(
            false, a[mi], false, b[ni], (short)0, c[mi][ni], false, false);
  }

  // Epilogue: + b1 (guarded for H pad), store C.
  // C layout: VGPR v -> row (v + 8*(lane>=16)), col = lane%16.
  const int rOff = lHi * 8;
#pragma unroll
  for (int ni = 0; ni < 4; ++ni) {
    int col = nBase + ni * 16 + l16;
    float bias = (col < H) ? b1[col] : 0.0f;
#pragma unroll
    for (int mi = 0; mi < 2; ++mi) {
      int m0 = mBase + mi * 16 + rOff;
#pragma unroll
      for (int v = 0; v < 8; ++v)
        cur1[(size_t)(m0 + v) * HPAD + col] = c[mi][ni][v] + bias;
    }
  }
}

// ---------------- Phase 2: 30-step LIF recurrence ----------------
// One wave32 per batch row. Lane L owns h = L + 32*j, j=0..31 (coalesced).
// cur1 pad columns are exactly 0 (zero W1 rows, no bias) -> never spike;
// W2 pad entries forced to 0 in LDS, so pad lanes contribute nothing.
__global__ __launch_bounds__(256)
void snn_timeloop_kernel(const float* __restrict__ cur1,
                         const float* __restrict__ W2,
                         const float* __restrict__ b2,
                         float* __restrict__ spk,
                         float* __restrict__ mem) {
  __shared__ float sW2[2][HPAD];
  for (int i = threadIdx.x; i < HPAD; i += 256) {
    sW2[0][i] = (i < H) ? W2[i]     : 0.0f;
    sW2[1][i] = (i < H) ? W2[H + i] : 0.0f;
  }
  __syncthreads();

  const int lane = threadIdx.x & 31;
  const int wave = threadIdx.x >> 5;
  const int row  = blockIdx.x * 8 + wave;

  float cur[32], m1[32];
  const float* cr = cur1 + (size_t)row * HPAD;
#pragma unroll
  for (int j = 0; j < 32; ++j) {
    cur[j] = cr[lane + j * 32];
    m1[j]  = 0.0f;
  }
  float m2a = 0.0f, m2b = 0.0f;
  const float b2a = b2[0], b2b = b2[1];

  for (int t = 0; t < TSTEPS; ++t) {
    float a0 = 0.0f, a1 = 0.0f;
#pragma unroll
    for (int j = 0; j < 32; ++j) {
      int   h  = lane + j * 32;
      float mo = m1[j];
      float r  = (mo > THR) ? THR : 0.0f;     // subtract-reset uses incoming mem
      float mn = fmaf(BETA, mo, cur[j]) - r;
      m1[j] = mn;
      float s  = (mn > THR) ? 1.0f : 0.0f;    // spike from updated mem
      a0 = fmaf(s, sW2[0][h], a0);
      a1 = fmaf(s, sW2[1][h], a1);
    }
    // wave32 butterfly all-reduce (warpSize == 32 on gfx1250)
#pragma unroll
    for (int off = 16; off >= 1; off >>= 1) {
      a0 += __shfl_xor(a0, off, 32);
      a1 += __shfl_xor(a1, off, 32);
    }
    float r0 = (m2a > THR) ? THR : 0.0f;
    float r1 = (m2b > THR) ? THR : 0.0f;
    m2a = fmaf(BETA, m2a, a0 + b2a) - r0;
    m2b = fmaf(BETA, m2b, a1 + b2b) - r1;
    if (lane == 0) {
      float s0 = (m2a > THR) ? 1.0f : 0.0f;
      float s1 = (m2b > THR) ? 1.0f : 0.0f;
      size_t o = ((size_t)t * BROWS + row) * 2;
      *(float2*)(spk + o) = make_float2(s0, s1);
      *(float2*)(mem + o) = make_float2(m2a, m2b);
    }
  }
}

// ---------------- launcher ----------------
extern "C" void kernel_launch(void* const* d_in, const int* in_sizes, int n_in,
                              void* d_out, int out_size, void* d_ws, size_t ws_size,
                              hipStream_t stream) {
  const float* x  = (const float*)d_in[0];   // [16384, 1024]
  const float* W1 = (const float*)d_in[1];   // [1000, 1024]
  const float* b1 = (const float*)d_in[2];   // [1000]
  const float* W2 = (const float*)d_in[3];   // [2, 1000]
  const float* b2 = (const float*)d_in[4];   // [2]
  float* out = (float*)d_out;                // spk [30,16384,2] ++ mem [30,16384,2]

  char* ws = (char*)d_ws;
  const size_t xb_bytes  = (size_t)BROWS * DIN * 2;   // 33,554,432
  const size_t w1b_bytes = (size_t)HPAD  * DIN * 2;   //  2,097,152
  __bf16* xb   = (__bf16*)ws;
  __bf16* w1b  = (__bf16*)(ws + xb_bytes);
  float*  cur1 = (float*)(ws + xb_bytes + w1b_bytes); // [16384, 1024] fp32

  float* spk = out;
  float* mem = out + (size_t)TSTEPS * BROWS * 2;

  // Phase 0: bf16 conversions (4 elems/thread)
  cvt_x_kernel <<<(BROWS * DIN) / (256 * 4), 256, 0, stream>>>(x, xb);
  cvt_w1_kernel<<<(HPAD  * DIN) / (256 * 4), 256, 0, stream>>>(W1, w1b);

  // Phase 1: WMMA GEMM, block tile 128x128
  dim3 ggrid(HPAD / 128, BROWS / 128);   // (8, 128)
  gemm_bf16_wmma_kernel<<<ggrid, 256, 0, stream>>>(xb, w1b, b1, cur1);

  // Phase 2: recurrence, 8 rows (waves) per block
  snn_timeloop_kernel<<<BROWS / 8, 256, 0, stream>>>(cur1, W2, b2, spk, mem);
}
